// SimpleSimProxy_89996744720969
// MI455X (gfx1250) — compile-verified, matrix operand order checked by reading, not run on hardware
//
#include <hip/hip_runtime.h>

#define BB 256
#define RR 128
#define LL 24
#define SS 1024
#define EE 1024

typedef __attribute__((ext_vector_type(2)))  float    v2f;
typedef __attribute__((ext_vector_type(8)))  float    v8f;
typedef __attribute__((ext_vector_type(16))) _Float16 v16h;

#if __has_builtin(__builtin_amdgcn_wmma_f32_16x16x4_f32)
#define HAVE_F32X4 1
#else
#define HAVE_F32X4 0
#endif

// ---------------------------------------------------------------------------
// Kernel 1: dedup one-hot histogram -> stop_counts (B, S) as float
// one block per batch row b; 128 threads, one route per thread.
// ---------------------------------------------------------------------------
__global__ __launch_bounds__(128) void k_hist(const int* __restrict__ stops,
                                              float* __restrict__ cnt) {
  __shared__ int hist[SS];
  const int b = blockIdx.x;
  for (int i = threadIdx.x; i < SS; i += 128) hist[i] = 0;
  __syncthreads();

  const int r = threadIdx.x;
  int st[LL];
  const int* p = stops + ((size_t)(b * RR + r)) * LL;
#pragma unroll
  for (int l = 0; l < LL; ++l) st[l] = p[l];
#pragma unroll
  for (int l = 0; l < LL; ++l) {
    bool dup = false;
#pragma unroll
    for (int m = 0; m < LL; ++m)
      if (m < l && st[m] == st[l]) dup = true;
    if (!dup) atomicAdd(&hist[st[l]], 1);
  }
  __syncthreads();
  for (int i = threadIdx.x; i < SS; i += 128)
    cnt[(size_t)b * SS + i] = (float)hist[i];
}

// ---------------------------------------------------------------------------
// Kernel 2: per-column mean / inv-std over batch axis (biased var, eps=1e-5)
// ---------------------------------------------------------------------------
__global__ void k_stats(const float* __restrict__ cnt, float* __restrict__ mean,
                        float* __restrict__ istd) {
  const int s = blockIdx.x * blockDim.x + threadIdx.x;
  float sum = 0.0f, sq = 0.0f;
  for (int b = 0; b < BB; ++b) {
    const float x = cnt[(size_t)b * SS + s];
    sum += x;
    sq += x * x;
  }
  const float m = sum * (1.0f / BB);
  const float v = sq * (1.0f / BB) - m * m;
  mean[s] = m;
  istd[s] = rsqrtf(v + 1e-5f);
}

// ---------------------------------------------------------------------------
// Kernel 3: normed = (cnt - mean) * invstd * gamma + beta
// ---------------------------------------------------------------------------
__global__ void k_norm(const float* __restrict__ cnt, const float* __restrict__ mean,
                       const float* __restrict__ istd, const float* __restrict__ gamma,
                       const float* __restrict__ beta, float* __restrict__ normed) {
  const int i = blockIdx.x * blockDim.x + threadIdx.x;
  const int s = i & (SS - 1);
  normed[i] = (cnt[i] - mean[s]) * istd[s] * gamma[s] + beta[s];
}

// ---------------------------------------------------------------------------
// Kernel 4: generic WMMA GEMM  C = act(A @ B + bias)
// A: MxK (lda=K), B: KxN (row stride ldb), C: MxN (ldc=N).
// Block = 128 threads (4 waves), tile 64x64, K-step 64.
// Wave w owns rows [w*16, w*16+16) of the block tile -> 4 C-fragments.
// ---------------------------------------------------------------------------
#define TM 64
#define TN 64
#define TK 64
#define LDT 68  // LDS row stride (floats): 16B-aligned float4 stores, staggered banks

__global__ __launch_bounds__(128) void k_gemm(const float* __restrict__ A,
                                              const float* __restrict__ Bm,
                                              const float* __restrict__ bias,
                                              float* __restrict__ C, int M, int N,
                                              int K, int ldb, int do_lrelu) {
  __shared__ float As[TM * LDT];
  __shared__ float Bs[TK * LDT];
  const int t = threadIdx.x;
  const int wave = t >> 5;
  const int lane = t & 31;
  const int lm = lane & 15;
  const int lh = lane >> 4;
  const int n0 = blockIdx.x * TN;
  const int m0 = blockIdx.y * TM;

  v8f acc[4] = {};

  const int ktiles = K / TK;
  for (int kt = 0; kt < ktiles; ++kt) {
    // Stage A tile (always in-bounds: M,K multiples of 64)
#pragma unroll
    for (int i = 0; i < 8; ++i) {
      const int idx = t + i * 128;
      const int row = idx >> 4;
      const int c4 = (idx & 15) << 2;
      const float4 av =
          *(const float4*)(A + (size_t)(m0 + row) * K + kt * TK + c4);
      *(float4*)(&As[row * LDT + c4]) = av;
    }
    // Stage B tile with N bounds (zero-fill OOB columns -> WMMA stays uniform)
#pragma unroll
    for (int i = 0; i < 8; ++i) {
      const int idx = t + i * 128;
      const int row = idx >> 4;
      const int c4 = (idx & 15) << 2;
      const int gn = n0 + c4;
      const float* src = Bm + (size_t)(kt * TK + row) * ldb + gn;
      float4 bv;
      if (gn + 3 < N) {
        bv = *(const float4*)src;
      } else {
        bv.x = (gn + 0 < N) ? src[0] : 0.0f;
        bv.y = (gn + 1 < N) ? src[1] : 0.0f;
        bv.z = (gn + 2 < N) ? src[2] : 0.0f;
        bv.w = (gn + 3 < N) ? src[3] : 0.0f;
      }
      *(float4*)(&Bs[row * LDT + c4]) = bv;
    }
    __syncthreads();

#if HAVE_F32X4
    // Exact fp32 path: V_WMMA_F32_16X16X4_F32
    // A frag (16x4): lane m=lm, VGPR0/1 = K{2*lh, 2*lh+1}
#pragma unroll
    for (int kk = 0; kk < 16; ++kk) {
      const int kb = kk * 4 + lh * 2;
      v2f a;
      a.x = As[(wave * 16 + lm) * LDT + kb];
      a.y = As[(wave * 16 + lm) * LDT + kb + 1];
#pragma unroll
      for (int nt = 0; nt < 4; ++nt) {
        v2f bf;
        bf.x = Bs[kb * LDT + nt * 16 + lm];
        bf.y = Bs[(kb + 1) * LDT + nt * 16 + lm];
        acc[nt] = __builtin_amdgcn_wmma_f32_16x16x4_f32(
            false, a, false, bf, (short)0, acc[nt], false, false);
      }
    }
#else
    // Fallback: f16 inputs, f32 accumulate (codegen-confirmed builtin)
#pragma unroll
    for (int kk = 0; kk < 2; ++kk) {
      v16h a;
#pragma unroll
      for (int v = 0; v < 8; ++v) {
        const int kq = kk * 32 + (v & 3) * 2 + (v >> 2) * 16 + lh * 8;
        a[2 * v] = (_Float16)As[(wave * 16 + lm) * LDT + kq];
        a[2 * v + 1] = (_Float16)As[(wave * 16 + lm) * LDT + kq + 1];
      }
#pragma unroll
      for (int nt = 0; nt < 4; ++nt) {
        v16h bf;
#pragma unroll
        for (int v = 0; v < 8; ++v) {
          const int kq = kk * 32 + (v & 3) * 2 + (v >> 2) * 16 + lh * 8;
          bf[2 * v] = (_Float16)Bs[kq * LDT + nt * 16 + lm];
          bf[2 * v + 1] = (_Float16)Bs[(kq + 1) * LDT + nt * 16 + lm];
        }
        acc[nt] = __builtin_amdgcn_wmma_f32_16x16x32_f16(
            false, a, false, bf, (short)0, acc[nt], false, false);
      }
    }
#endif
    __syncthreads();
  }

  // Epilogue: C/D layout -> VGPR v: lanes 0-15 = row v, lanes 16-31 = row v+8
#pragma unroll
  for (int nt = 0; nt < 4; ++nt) {
    const int gn = n0 + nt * 16 + lm;
    if (gn < N) {
      const float bv = bias[gn];
#pragma unroll
      for (int v = 0; v < 8; ++v) {
        const int gm = m0 + wave * 16 + lh * 8 + v;
        float val = acc[nt][v] + bv;
        if (do_lrelu) val = val > 0.0f ? val : 0.01f * val;
        C[(size_t)gm * N + gn] = val;
      }
    }
  }
}

// ---------------------------------------------------------------------------
// Kernel 5: split outs (256 x 1025) -> global_preds (d_out[0:256]) and
// g = lrelu(outs[:,1:]) (256 x 1024)
// ---------------------------------------------------------------------------
__global__ void k_split(const float* __restrict__ outs, float* __restrict__ out,
                        float* __restrict__ g) {
  const int i = blockIdx.x * blockDim.x + threadIdx.x;
  if (i < BB) out[i] = outs[(size_t)i * (EE + 1)];
  const int b = i / EE;
  const int j = i - b * EE;
  const float v = outs[(size_t)b * (EE + 1) + 1 + j];
  g[i] = v > 0.0f ? v : 0.01f * v;
}

// ---------------------------------------------------------------------------
// Kernel 6: fused route head. One wave per (b,r):
//   rh = lrelu(P[b] + sum_{distinct s in route} Wr0[E+s, :])   (br0 folded in P)
//   route_pred = rh . Wr1 + br1
// Dedup via shfl + ballot/popc compaction into per-wave LDS list.
// ---------------------------------------------------------------------------
__global__ __launch_bounds__(256) void k_route(const int* __restrict__ stops,
                                               const float* __restrict__ P,
                                               const float* __restrict__ Wr0,
                                               const float* __restrict__ Wr1,
                                               const float* __restrict__ br1,
                                               float* __restrict__ out) {
  __shared__ int sstop[8][32];
  __shared__ int scount[8];
  const int wave = threadIdx.x >> 5;
  const int lane = threadIdx.x & 31;
  const int pair = blockIdx.x * 8 + wave;  // b*R + r
  const int b = pair >> 7;

  int s = 0;
  if (lane < LL) s = stops[(size_t)pair * LL + lane];
  bool valid = (lane < LL);
  for (int m = 0; m < LL; ++m) {
    const int sm = __shfl(s, m);
    if (m < lane && sm == s) valid = false;  // keep first occurrence only
  }
  const unsigned mask32 = (unsigned)__ballot(valid);
  const int pos = __popc(mask32 & ((1u << lane) - 1u));
  if (valid) sstop[wave][pos] = s;
  if (lane == 0) scount[wave] = __popc(mask32);
  __syncthreads();

  const int nstop = scount[wave];
  const float* Prow = P + (size_t)b * EE;
  float dot = 0.0f;
#pragma unroll
  for (int ch = 0; ch < 8; ++ch) {
    const int j = ch * 128 + lane * 4;
    float4 acc = *(const float4*)(Prow + j);
    for (int t2 = 0; t2 < nstop; ++t2) {
      const int st = sstop[wave][t2];
      const float4 wv = *(const float4*)(Wr0 + (size_t)(EE + st) * EE + j);
      acc.x += wv.x;
      acc.y += wv.y;
      acc.z += wv.z;
      acc.w += wv.w;
    }
    acc.x = acc.x > 0.0f ? acc.x : 0.01f * acc.x;
    acc.y = acc.y > 0.0f ? acc.y : 0.01f * acc.y;
    acc.z = acc.z > 0.0f ? acc.z : 0.01f * acc.z;
    acc.w = acc.w > 0.0f ? acc.w : 0.01f * acc.w;
    const float4 w1 = *(const float4*)(Wr1 + j);
    dot += acc.x * w1.x + acc.y * w1.y + acc.z * w1.z + acc.w * w1.w;
  }
#pragma unroll
  for (int off = 16; off >= 1; off >>= 1) dot += __shfl_xor(dot, off);
  if (lane == 0) out[BB + pair] = dot + br1[0];
}

// ---------------------------------------------------------------------------
extern "C" void kernel_launch(void* const* d_in, const int* in_sizes, int n_in,
                              void* d_out, int out_size, void* d_ws, size_t ws_size,
                              hipStream_t stream) {
  (void)in_sizes; (void)n_in; (void)out_size; (void)ws_size;
  const int* route_stops = (const int*)d_in[0];
  const float* gamma = (const float*)d_in[1];
  const float* beta = (const float*)d_in[2];
  const float* W0 = (const float*)d_in[3];
  const float* b0 = (const float*)d_in[4];
  const float* W1 = (const float*)d_in[5];
  const float* b1 = (const float*)d_in[6];
  const float* W2 = (const float*)d_in[7];
  const float* b2 = (const float*)d_in[8];
  const float* Wb = (const float*)d_in[9];
  const float* bbv = (const float*)d_in[10];
  const float* Wr0 = (const float*)d_in[11];
  const float* br0 = (const float*)d_in[12];
  const float* Wr1 = (const float*)d_in[13];
  const float* br1 = (const float*)d_in[14];
  float* out = (float*)d_out;

  float* w = (float*)d_ws;
  float* cnt = w;                       // B*S
  float* meanp = cnt + BB * SS;         // S
  float* istd = meanp + SS;             // S
  float* normed = istd + SS;            // B*E
  float* h0v = normed + BB * EE;        // B*E
  float* h1v = h0v + BB * EE;           // B*E
  float* h2v = h1v + BB * EE;           // B*E
  float* outs = h2v + BB * EE;          // B*(E+1)
  float* gbuf = outs + BB * (EE + 1);   // B*E
  float* Pbuf = gbuf + BB * EE;         // B*E   (~8.4 MB total)

  k_hist<<<BB, 128, 0, stream>>>(route_stops, cnt);
  k_stats<<<SS / 256, 256, 0, stream>>>(cnt, meanp, istd);
  k_norm<<<(BB * SS) / 256, 256, 0, stream>>>(cnt, meanp, istd, gamma, beta, normed);

  auto gemm = [&](const float* Aa, const float* Bb, const float* bias, float* Cc,
                  int N, int ldb, int lrelu) {
    dim3 grid((N + TN - 1) / TN, BB / TM);
    k_gemm<<<grid, 128, 0, stream>>>(Aa, Bb, bias, Cc, BB, N, EE, ldb, lrelu);
  };
  gemm(normed, W0, b0, h0v, EE, EE, 1);
  gemm(h0v, W1, b1, h1v, EE, EE, 1);
  gemm(h1v, W2, b2, h2v, EE, EE, 1);
  gemm(h2v, Wb, bbv, outs, EE + 1, EE + 1, 0);
  k_split<<<(BB * EE) / 256, 256, 0, stream>>>(outs, out, gbuf);
  // P = g @ Wr0[:E, :] + br0  (r-invariant half of the route head)
  gemm(gbuf, Wr0, br0, Pbuf, EE, EE, 0);
  k_route<<<(BB * RR) / 8, 256, 0, stream>>>(route_stops, Pbuf, Wr0, Wr1, br1, out);
}
